// MinibatchDiscrimination_16758962389448
// MI455X (gfx1250) — matrix operand order repaired and checked by reading, not compiled, not run on hardware
//
#include <hip/hip_runtime.h>
#include <stdint.h>

typedef _Float16 half_t;
typedef __attribute__((ext_vector_type(16))) _Float16 v16h;
typedef __attribute__((ext_vector_type(8)))  float    v8f;
typedef int v4i_vs __attribute__((vector_size(16)));   // matches builtin's pointee type

#define B_DIM 256
#define F_DIM 1024
#define O_DIM 128
#define I_DIM 16
#define N_DIM (O_DIM * I_DIM)      // 2048
#define OUT_STRIDE (F_DIM + O_DIM) // 1152
#define KT_STEPS (F_DIM / 32)      // 32 k-tiles of 32

// d_ws layout (bytes): [0, 2M) Mws f32 | [2M, 2.5M) Apack f16 | [2.5M, 6.5M) Bpack f16
#define WS_APACK_OFF (2u * 1024u * 1024u)
#define WS_BPACK_OFF (2u * 1024u * 1024u + 512u * 1024u)

#if defined(__gfx1250__) && __has_builtin(__builtin_amdgcn_global_load_async_to_lds_b128)
#define HAVE_ASYNC_LDS 1
#endif

// Fragment indexing: frag(tile, kt) is 32 lanes x 16 halves (32B/lane, contiguous per lane)
__device__ __forceinline__ size_t frag_off(int tile, int kt, int lane) {
    return ((size_t)(tile * KT_STEPS + kt) * 32 + lane) * 16;
}

// ---------------------------------------------------------------------------
// Pack A: x[256,1024] f32 -> f16 fragments in WMMA A-layout (16-bit 16x32):
//   VGPR v holds K = 2v (+8 if v>=4) (+8 for lanes 16-31), pair (K, K+1)
// ---------------------------------------------------------------------------
__global__ __launch_bounds__(32)
void mbd_pack_a(const float* __restrict__ x, half_t* __restrict__ Apack) {
    const int kt = blockIdx.x, jt = blockIdx.y;
    const int lane = threadIdx.x, l = lane & 15, hi = lane >> 4;
    const float* xrow = x + (size_t)(jt * 16 + l) * F_DIM + kt * 32;
    v16h a;
#pragma unroll
    for (int v = 0; v < 8; ++v) {
        const int ka = 2 * v + (v >= 4 ? 8 : 0) + hi * 8;
        const float2 xv = *(const float2*)(xrow + ka);
        a[2 * v]     = (half_t)xv.x;
        a[2 * v + 1] = (half_t)xv.y;
    }
    *(v16h*)(Apack + frag_off(jt, kt, lane)) = a;
}

// ---------------------------------------------------------------------------
// Pack B: T[1024,2048] f32 -> f16 fragments in WMMA B-layout (16-bit 32x16):
//   VGPR v holds K = 2v (+8 if v>=4) (+8 for lanes 16-31), pair (K, K+1); N = lane&15
// ---------------------------------------------------------------------------
__global__ __launch_bounds__(32)
void mbd_pack_b(const float* __restrict__ T, half_t* __restrict__ Bpack) {
    const int kt = blockIdx.x, ot = blockIdx.y;
    const int lane = threadIdx.x, l = lane & 15, hi = lane >> 4;
    const int n = ot * 16 + l;
    v16h b;
#pragma unroll
    for (int v = 0; v < 8; ++v) {
        const int k = kt * 32 + 2 * v + (v >= 4 ? 8 : 0) + hi * 8;
        b[2 * v]     = (half_t)T[(size_t)k * N_DIM + n];
        b[2 * v + 1] = (half_t)T[(size_t)(k + 1) * N_DIM + n];
    }
    *(v16h*)(Bpack + frag_off(ot, kt, lane)) = b;
}

// ---------------------------------------------------------------------------
// GEMM: one wave32 computes a 2x2 block of 16x16 tiles (32 rows x 32 cols).
// Inner loop: 8 x global_load_b128 + 4 x v_wmma_f32_16x16x32_f16.
// Output scattered to Mws[o][j][k] (o = N/16, k = N%16).
// ---------------------------------------------------------------------------
__global__ __launch_bounds__(32)
void mbd_gemm_wmma(const half_t* __restrict__ Apack,
                   const half_t* __restrict__ Bpack,
                   float* __restrict__ Mws) {
    const int ot0 = blockIdx.x * 2;  // 0..126 step 2
    const int jt0 = blockIdx.y * 2;  // 0..14  step 2
    const int lane = threadIdx.x, l = lane & 15, hi = lane >> 4;

    const half_t* pa0 = Apack + frag_off(jt0,     0, lane);
    const half_t* pa1 = Apack + frag_off(jt0 + 1, 0, lane);
    const half_t* pb0 = Bpack + frag_off(ot0,     0, lane);
    const half_t* pb1 = Bpack + frag_off(ot0 + 1, 0, lane);

    v8f acc00 = {}, acc01 = {}, acc10 = {}, acc11 = {};
    for (int kt = 0; kt < KT_STEPS; ++kt) {
        const v16h a0 = *(const v16h*)(pa0 + (size_t)kt * 32 * 16);
        const v16h a1 = *(const v16h*)(pa1 + (size_t)kt * 32 * 16);
        const v16h b0 = *(const v16h*)(pb0 + (size_t)kt * 32 * 16);
        const v16h b1 = *(const v16h*)(pb1 + (size_t)kt * 32 * 16);
        acc00 = __builtin_amdgcn_wmma_f32_16x16x32_f16(false, a0, false, b0, (short)0, acc00, false, false);
        acc01 = __builtin_amdgcn_wmma_f32_16x16x32_f16(false, a0, false, b1, (short)0, acc01, false, false);
        acc10 = __builtin_amdgcn_wmma_f32_16x16x32_f16(false, a1, false, b0, (short)0, acc10, false, false);
        acc11 = __builtin_amdgcn_wmma_f32_16x16x32_f16(false, a1, false, b1, (short)0, acc11, false, false);
    }

    // C/D layout: VGPR r -> M=r (lanes 0-15) / M=r+8 (lanes 16-31); N = l
#pragma unroll
    for (int r = 0; r < 8; ++r) {
        const int j0 = jt0 * 16 + r + 8 * hi;
        const int j1 = j0 + 16;
        Mws[(size_t)ot0       * (B_DIM * I_DIM) + j0 * I_DIM + l] = acc00[r];
        Mws[(size_t)(ot0 + 1) * (B_DIM * I_DIM) + j0 * I_DIM + l] = acc01[r];
        Mws[(size_t)ot0       * (B_DIM * I_DIM) + j1 * I_DIM + l] = acc10[r];
        Mws[(size_t)(ot0 + 1) * (B_DIM * I_DIM) + j1 * I_DIM + l] = acc11[r];
    }
}

// ---------------------------------------------------------------------------
// Similarity: o_feat[i,o] = sum_{j!=i} exp(-sum_k |M[j,o,k]-M[i,o,k]|)
// Block per o; M[:,o,:] (16KB) staged into LDS via async-to-LDS when available.
// ---------------------------------------------------------------------------
__global__ __launch_bounds__(256)
void mbd_sim(const float* __restrict__ Mws, float* __restrict__ out) {
    const int o = blockIdx.x;
    const int i = threadIdx.x;

    __shared__ float lds[B_DIM * I_DIM]; // 16 KB

    const float* src = Mws + (size_t)o * (B_DIM * I_DIM);
#if HAVE_ASYNC_LDS
    {
        // builtin expects (int4 addrspace(1)*, int4 addrspace(3)*, imm offset, imm cpol)
        auto* g1 = (__attribute__((address_space(1))) v4i_vs*)(uintptr_t)src;
        auto* l3 = (__attribute__((address_space(3))) v4i_vs*)(uint32_t)(uintptr_t)&lds[0];
#pragma unroll
        for (int it = 0; it < 4; ++it) {
            const int idx = it * 256 + threadIdx.x; // one b128 per thread per issue
            __builtin_amdgcn_global_load_async_to_lds_b128(g1 + idx, l3 + idx, 0, 0);
        }
#if __has_builtin(__builtin_amdgcn_s_wait_asynccnt)
        __builtin_amdgcn_s_wait_asynccnt(0);
#else
        asm volatile("s_wait_asynccnt 0" ::: "memory");
#endif
    }
#else
    for (int idx = threadIdx.x; idx < B_DIM * I_DIM; idx += 256)
        lds[idx] = src[idx];
#endif
    __syncthreads();

    const float4* l4 = (const float4*)lds;
    const float4 mi0 = l4[i * 4 + 0], mi1 = l4[i * 4 + 1];
    const float4 mi2 = l4[i * 4 + 2], mi3 = l4[i * 4 + 3];

    float acc = 0.0f;
    for (int j = 0; j < B_DIM; ++j) {
        const float4 r0 = l4[j * 4 + 0], r1 = l4[j * 4 + 1];
        const float4 r2 = l4[j * 4 + 2], r3 = l4[j * 4 + 3];
        float d = fabsf(r0.x - mi0.x) + fabsf(r0.y - mi0.y) + fabsf(r0.z - mi0.z) + fabsf(r0.w - mi0.w);
        d += fabsf(r1.x - mi1.x) + fabsf(r1.y - mi1.y) + fabsf(r1.z - mi1.z) + fabsf(r1.w - mi1.w);
        d += fabsf(r2.x - mi2.x) + fabsf(r2.y - mi2.y) + fabsf(r2.z - mi2.z) + fabsf(r2.w - mi2.w);
        d += fabsf(r3.x - mi3.x) + fabsf(r3.y - mi3.y) + fabsf(r3.z - mi3.z) + fabsf(r3.w - mi3.w);
        acc += (j == i) ? 0.0f : __expf(-d);
    }

    out[(size_t)i * OUT_STRIDE + F_DIM + o] = acc;
}

// ---------------------------------------------------------------------------
// Copy x into out[:, 0:1024]
// ---------------------------------------------------------------------------
__global__ __launch_bounds__(256)
void mbd_copy_x(const float* __restrict__ x, float* __restrict__ out) {
    const int i = blockIdx.x;
    const float4* src = (const float4*)(x + (size_t)i * F_DIM);
    float4*       dst = (float4*)(out + (size_t)i * OUT_STRIDE);
    dst[threadIdx.x] = src[threadIdx.x];
}

// ---------------------------------------------------------------------------
extern "C" void kernel_launch(void* const* d_in, const int* in_sizes, int n_in,
                              void* d_out, int out_size, void* d_ws, size_t ws_size,
                              hipStream_t stream) {
    const float* x = (const float*)d_in[0];   // [256, 1024] f32
    const float* T = (const float*)d_in[1];   // [1024, 128, 16] f32
    float* out = (float*)d_out;               // [256, 1152] f32

    float*  Mws   = (float*)d_ws;                              // 2 MB
    half_t* Apack = (half_t*)((char*)d_ws + WS_APACK_OFF);     // 512 KB
    half_t* Bpack = (half_t*)((char*)d_ws + WS_BPACK_OFF);     // 4 MB

    mbd_pack_a<<<dim3(KT_STEPS, B_DIM / 16), 32, 0, stream>>>(x, Apack);
    mbd_pack_b<<<dim3(KT_STEPS, O_DIM), 32, 0, stream>>>(T, Bpack);
    mbd_gemm_wmma<<<dim3(O_DIM / 2, B_DIM / 32), 32, 0, stream>>>(Apack, Bpack, Mws);
    mbd_copy_x<<<B_DIM, 256, 0, stream>>>(x, out);
    mbd_sim<<<O_DIM, 256, 0, stream>>>(Mws, out);
}
